// RNNNetwork_81905026335347
// MI455X (gfx1250) — matrix operand order repaired and checked by reading, not compiled, not run on hardware
//
#include <hip/hip_runtime.h>
#include <hip/hip_bf16.h>

// ---------------------------------------------------------------------------
// Fused 2-layer tanh-RNN for (B,T,I,H) = (256, 336, 7, 512) on gfx1250.
// One workgroup per 16 batch rows runs the whole time loop; all GEMMs are
// v_wmma_f32_16x16x32_f16 with f32 accumulation.
// ---------------------------------------------------------------------------

#define RNN_B 256
#define RNN_T 336
#define RNN_I 7
#define RNN_H 512
#define LSTR  520          // LDS row stride in halves (1040B rows -> clean banks)

typedef __attribute__((ext_vector_type(16))) _Float16 v16h;
typedef __attribute__((ext_vector_type(8)))  _Float16 v8h;
typedef __attribute__((ext_vector_type(8)))  float    v8f;

// --------------------------- weight prep kernel ----------------------------
__global__ __launch_bounds__(256) void rnn_prep_kernel(
    const float* __restrict__ wh0, const float* __restrict__ wi1,
    const float* __restrict__ wh1,
    const float* __restrict__ bih0, const float* __restrict__ bhh0,
    const float* __restrict__ bih1, const float* __restrict__ bhh1,
    _Float16* __restrict__ wh0h, _Float16* __restrict__ wi1h,
    _Float16* __restrict__ wh1h, float* __restrict__ b0,
    float* __restrict__ b1) {
  int i = blockIdx.x * 256 + threadIdx.x;
  if (i < RNN_H * RNN_H) {
    wh0h[i] = (_Float16)wh0[i];
    wi1h[i] = (_Float16)wi1[i];
    wh1h[i] = (_Float16)wh1[i];
  }
  if (i < RNN_H) {
    b0[i] = bih0[i] + bhh0[i];
    b1[i] = bih1[i] + bhh1[i];
  }
}

// --------------- K=512 GEMM accumulate: C += A(lds) x B(Wᵀ) ----------------
// A: 16 x 512 f16 tile in LDS (row stride LSTR).
// W: 512 x 512 f16 row-major global; B[k][n] = W[n][k], so a WMMA B-column is
// a contiguous row of W.
__device__ __forceinline__ void gemm_acc(v8f c[4],
                                         const _Float16* __restrict__ hsrc,
                                         const _Float16* __restrict__ W,
                                         int lane, int nbase) {
  const int m  = lane & 15;
  const int kh = lane >> 4;
  const _Float16* arow = hsrc + m * LSTR;
#pragma unroll 4
  for (int k0 = 0; k0 < RNN_H; k0 += 32) {
    union { v16h v; v8h h[2]; } a;
    a.h[0] = *(const v8h*)(arow + k0 + 16 * kh);       // K = k0+16kh .. +8
    a.h[1] = *(const v8h*)(arow + k0 + 16 * kh + 8);   // K = k0+16kh+8 .. +8
#pragma unroll
    for (int j = 0; j < 4; ++j) {
      const _Float16* brow = W + (nbase + 16 * j + m) * RNN_H + k0 + 16 * kh;
      v16h b = *(const v16h*)brow;                     // 32B contiguous
      c[j] = __builtin_amdgcn_wmma_f32_16x16x32_f16(
          false, a.v, false, b, (short)0, c[j], false, false);
    }
  }
}

// ------------------------------ main kernel --------------------------------
__global__ __launch_bounds__(256) void rnn_fused_kernel(
    const float* __restrict__ x,        // (B,T,I) f32
    const float* __restrict__ wih0g,    // (H,I)   f32
    const _Float16* __restrict__ wh0h,  // (H,H)   f16
    const _Float16* __restrict__ wi1h,  // (H,H)   f16
    const _Float16* __restrict__ wh1h,  // (H,H)   f16
    const float* __restrict__ b0g,      // (H)     f32  (b_ih0+b_hh0)
    const float* __restrict__ b1g,      // (H)     f32  (b_ih1+b_hh1)
    const float* __restrict__ fcw,      // (H)     f32
    const float* __restrict__ fcb,      // (1)     f32
    float* __restrict__ out) {          // (B)     f32
  __shared__ _Float16 h0s[2][16 * LSTR];
  __shared__ _Float16 h1s[2][16 * LSTR];
  __shared__ float    hfinal[16 * LSTR];   // f32 copy of last h1
  __shared__ float    xts[16 * 8];         // x tile for current t (pad 8)
  __shared__ float    wih0s[RNN_H * 8];    // W_ih0 rows, pad to 8
  __shared__ float    b0s[RNN_H];
  __shared__ float    b1s[RNN_H];

  const int tid   = threadIdx.x;
  const int lane  = tid & 31;
  const int wave  = tid >> 5;          // 0..7 -> columns [64w, 64w+64)
  const int nbase = wave * 64;
  const int m     = lane & 15;
  const int kh    = lane >> 4;
  const int ncol  = lane & 15;
  const int bbase = blockIdx.x * 16;   // 16 batch rows per workgroup

  // ---- one-time setup: zero h state, stage W_ih0 / biases into LDS ----
  for (int i = tid; i < 16 * LSTR; i += 256) {
    h0s[0][i] = (_Float16)0.0f;
    h1s[0][i] = (_Float16)0.0f;
  }
  for (int i = tid; i < RNN_H * RNN_I; i += 256) {
    int n = i / RNN_I, q = i % RNN_I;
    wih0s[n * 8 + q] = wih0g[i];
  }
  for (int i = tid; i < RNN_H; i += 256) {
    wih0s[i * 8 + 7] = 0.0f;
    b0s[i] = b0g[i];
    b1s[i] = b1g[i];
  }
  __syncthreads();

  for (int t = 0; t < RNN_T; ++t) {
    const int pr = t & 1;        // read buffer
    const int pw = pr ^ 1;       // write buffer

    // ---- stage x[bbase..bbase+15, t, 0..6] ----
    if (tid < 128) {
      int row = tid >> 3, q = tid & 7;
      xts[tid] = (q < RNN_I)
                     ? x[(size_t)(bbase + row) * RNN_T * RNN_I + t * RNN_I + q]
                     : 0.0f;
    }
    __syncthreads();

    // ================= phase 1: h0 = tanh(pre0 + h0·W_hh0ᵀ) ==============
    {
      v8f c[4];
#pragma unroll
      for (int j = 0; j < 4; ++j) {
        const int n = nbase + 16 * j + ncol;
        const float bv = b0s[n];
#pragma unroll
        for (int r = 0; r < 8; ++r) {
          const int row = r + 8 * kh;
          float acc = bv;
#pragma unroll
          for (int q = 0; q < RNN_I; ++q)
            acc += xts[row * 8 + q] * wih0s[n * 8 + q];
          c[j][r] = acc;
        }
      }
      gemm_acc(c, h0s[pr], wh0h, lane, nbase);
#pragma unroll
      for (int j = 0; j < 4; ++j) {
        const int n = nbase + 16 * j + ncol;
#pragma unroll
        for (int r = 0; r < 8; ++r) {
          const int row = r + 8 * kh;
          h0s[pw][row * LSTR + n] = (_Float16)tanhf(c[j][r]);
        }
      }
    }
    __syncthreads();

    // ====== phase 2: h1 = tanh(b1 + h0_new·W_ih1ᵀ + h1·W_hh1ᵀ) ===========
    {
      v8f c[4];
#pragma unroll
      for (int j = 0; j < 4; ++j) {
        const float bv = b1s[nbase + 16 * j + ncol];
#pragma unroll
        for (int r = 0; r < 8; ++r) c[j][r] = bv;
      }
      gemm_acc(c, h0s[pw], wi1h, lane, nbase);
      gemm_acc(c, h1s[pr], wh1h, lane, nbase);
#pragma unroll
      for (int j = 0; j < 4; ++j) {
        const int n = nbase + 16 * j + ncol;
#pragma unroll
        for (int r = 0; r < 8; ++r) {
          const int row = r + 8 * kh;
          const float v = tanhf(c[j][r]);
          h1s[pw][row * LSTR + n] = (_Float16)v;
          if (t == RNN_T - 1) hfinal[row * LSTR + n] = v;   // full f32 last state
        }
      }
    }
    __syncthreads();
  }

  // ---- FC head: out[b] = hfinal[b,:]·fc_w + fc_b (f32) ----
  if (tid < 16) {
    float acc = fcb[0];
    for (int k = 0; k < RNN_H; ++k) acc += hfinal[tid * LSTR + k] * fcw[k];
    out[bbase + tid] = acc;
  }
}

// ------------------------------ launcher -----------------------------------
extern "C" void kernel_launch(void* const* d_in, const int* in_sizes, int n_in,
                              void* d_out, int out_size, void* d_ws, size_t ws_size,
                              hipStream_t stream) {
  const float* x     = (const float*)d_in[0];
  const float* W_ih0 = (const float*)d_in[1];
  const float* W_hh0 = (const float*)d_in[2];
  const float* b_ih0 = (const float*)d_in[3];
  const float* b_hh0 = (const float*)d_in[4];
  const float* W_ih1 = (const float*)d_in[5];
  const float* W_hh1 = (const float*)d_in[6];
  const float* b_ih1 = (const float*)d_in[7];
  const float* b_hh1 = (const float*)d_in[8];
  const float* fc_w  = (const float*)d_in[9];
  const float* fc_b  = (const float*)d_in[10];
  float* out = (float*)d_out;

  // workspace layout: 3 f16 weight matrices + 2 fused f32 bias vectors
  _Float16* wh0h = (_Float16*)d_ws;
  _Float16* wi1h = wh0h + RNN_H * RNN_H;
  _Float16* wh1h = wi1h + RNN_H * RNN_H;
  float*    b0   = (float*)(wh1h + RNN_H * RNN_H);
  float*    b1   = b0 + RNN_H;

  rnn_prep_kernel<<<(RNN_H * RNN_H + 255) / 256, 256, 0, stream>>>(
      W_hh0, W_ih1, W_hh1, b_ih0, b_hh0, b_ih1, b_hh1,
      wh0h, wi1h, wh1h, b0, b1);

  rnn_fused_kernel<<<RNN_B / 16, 256, 0, stream>>>(
      x, W_ih0, wh0h, wi1h, wh1h, b0, b1, fc_w, fc_b, out);
}
